// WindowAttention_29583734735087
// MI455X (gfx1250) — matrix operand order repaired and checked by reading, not compiled, not run on hardware
//
#include <hip/hip_runtime.h>
#include <hip/hip_bf16.h>
#include <stdint.h>

typedef __attribute__((ext_vector_type(16))) _Float16 v16h;
typedef __attribute__((ext_vector_type(8)))  float    v8f;

#define NT      49            // tokens per window
#define NP      64            // padded tokens
#define NHEADS  6
#define HD      32            // head dim
#define CDIM    192           // model dim
#define C3      576           // 3*CDIM (qkv row stride in f32)
#define TBL     169
#define ATT_SCALE 0.17677669529663687f   // 32^-0.5

union H8 { uint4 q; _Float16 h[8]; };

__device__ __forceinline__ v16h zero16() { v16h z = {}; return z; }

// D = A*B + C, 16x16x32 f16 -> f32 accum (gfx1250 WMMA)
__device__ __forceinline__ v8f wmma(v16h a, v16h b, v8f c) {
  return __builtin_amdgcn_wmma_f32_16x16x32_f16(false, a, false, b, (short)0, c,
                                                false, false);
}

// A-fragment (16x32 f16, M=lane&15) from an f32 row in global memory.
// VGPR0-7 hold K=koff..koff+15, koff=(lane>=16)*16.
__device__ __forceinline__ v16h a_frag_f32row(const float* __restrict__ prow, int lane) {
  const int koff = ((lane >> 4) << 4);
  const float4 x0 = *(const float4*)(prow + koff);
  const float4 x1 = *(const float4*)(prow + koff + 4);
  const float4 y0 = *(const float4*)(prow + 8 + koff);
  const float4 y1 = *(const float4*)(prow + 12 + koff);
  v16h f;
  f[0]  = (_Float16)x0.x; f[1]  = (_Float16)x0.y; f[2]  = (_Float16)x0.z; f[3]  = (_Float16)x0.w;
  f[4]  = (_Float16)x1.x; f[5]  = (_Float16)x1.y; f[6]  = (_Float16)x1.z; f[7]  = (_Float16)x1.w;
  f[8]  = (_Float16)y0.x; f[9]  = (_Float16)y0.y; f[10] = (_Float16)y0.z; f[11] = (_Float16)y0.w;
  f[12] = (_Float16)y1.x; f[13] = (_Float16)y1.y; f[14] = (_Float16)y1.z; f[15] = (_Float16)y1.w;
  return f;
}

// B-fragment (32x16 f16, N=lane&15): 16 contiguous K-values from global f32.
__device__ __forceinline__ v16h b_frag_f32row(const float* __restrict__ p16) {
  const float4 x0 = *(const float4*)(p16);
  const float4 x1 = *(const float4*)(p16 + 4);
  const float4 x2 = *(const float4*)(p16 + 8);
  const float4 x3 = *(const float4*)(p16 + 12);
  v16h f;
  f[0]  = (_Float16)x0.x; f[1]  = (_Float16)x0.y; f[2]  = (_Float16)x0.z; f[3]  = (_Float16)x0.w;
  f[4]  = (_Float16)x1.x; f[5]  = (_Float16)x1.y; f[6]  = (_Float16)x1.z; f[7]  = (_Float16)x1.w;
  f[8]  = (_Float16)x2.x; f[9]  = (_Float16)x2.y; f[10] = (_Float16)x2.z; f[11] = (_Float16)x2.w;
  f[12] = (_Float16)x3.x; f[13] = (_Float16)x3.y; f[14] = (_Float16)x3.z; f[15] = (_Float16)x3.w;
  return f;
}

// A-fragment from f16 LDS: two 16B chunks -> two ds_load_b128
__device__ __forceinline__ v16h a_frag_lds(const _Float16* base, int row, int stride,
                                           int koff) {
  const _Float16* p = base + row * stride + koff;
  H8 a, b;
  a.q = *(const uint4*)p;
  b.q = *(const uint4*)(p + 8);
  v16h f;
#pragma unroll
  for (int j = 0; j < 8; ++j) { f[j] = a.h[j]; f[8 + j] = b.h[j]; }
  return f;
}

// B-fragment from f16 LDS: 32 contiguous bytes -> two ds_load_b128
__device__ __forceinline__ v16h b_frag_lds(const _Float16* p16) {
  H8 a, b;
  a.q = *(const uint4*)p16;
  b.q = *(const uint4*)(p16 + 8);
  v16h f;
#pragma unroll
  for (int j = 0; j < 8; ++j) { f[j] = a.h[j]; f[8 + j] = b.h[j]; }
  return f;
}

__global__ __launch_bounds__(256) void swin_win_attn_fused(
    const float* __restrict__ qkv, const long long* __restrict__ rpi,
    const float* __restrict__ mask, const float* __restrict__ bias_table,
    const float* __restrict__ proj_w, const float* __restrict__ proj_b,
    float* __restrict__ out, int nw) {
  extern __shared__ char smem[];
  _Float16* sVt   = (_Float16*)smem;                              // [6][32][64]
  _Float16* sAttn = (_Float16*)(smem + NHEADS * HD * NP * 2);     // [6][64][64]
  _Float16* sX    = (_Float16*)(smem + NHEADS * HD * NP * 2 +
                                NHEADS * NP * NP * 2);            // [64][192]

  const int lane = threadIdx.x & 31;
  const int wave = threadIdx.x >> 5;
  const int b    = blockIdx.x;
  const int widx = b % nw;
  const float* qkv_b = qkv + (size_t)b * NT * C3;

  if (wave < NHEADS) {
    const int h = wave;
    _Float16* sVtH   = sVt + h * HD * NP;
    _Float16* sAttnH = sAttn + h * NP * NP;
    const int qoff = h * HD;
    const int koffg = CDIM + h * HD;
    const int voff = 2 * CDIM + h * HD;

    // ---- stage V transposed into LDS: sVt[d][t] (f16), zero-padded tokens.
    // Coalesced float4 reads along the head dim, transpose-scatter b16 stores.
    for (int lin = lane; lin < NP * (HD / 4); lin += 32) {
      const int t  = lin >> 3;          // token
      const int c4 = (lin & 7) << 2;    // dim group of 4
      float4 v = (t < NT) ? *(const float4*)(qkv_b + (size_t)t * C3 + voff + c4)
                          : make_float4(0.f, 0.f, 0.f, 0.f);
      sVtH[(c4 + 0) * NP + t] = (_Float16)v.x;
      sVtH[(c4 + 1) * NP + t] = (_Float16)v.y;
      sVtH[(c4 + 2) * NP + t] = (_Float16)v.z;
      sVtH[(c4 + 3) * NP + t] = (_Float16)v.w;
    }

    // ---- K^T B-fragments, straight from global (contiguous K rows)
    v16h kf[4];
    {
      const int nn = lane & 15;
      const int kb = ((lane >> 4) << 4);
#pragma unroll
      for (int nt = 0; nt < 4; ++nt) {
        const int t = nt * 16 + nn;
        kf[nt] = (t < NT) ? b_frag_f32row(qkv_b + (size_t)t * C3 + koffg + kb)
                          : zero16();
      }
    }

    const int n0 = lane & 15;
#pragma unroll
    for (int mt = 0; mt < 4; ++mt) {
      const int mq = mt * 16 + (lane & 15);
      v16h qa = (mq < NT) ? a_frag_f32row(qkv_b + (size_t)mq * C3 + qoff, lane)
                          : zero16();
      // ---- scores: one WMMA per 16x16 tile (K = head_dim = 32)
      v8f s[4];
#pragma unroll
      for (int nt = 0; nt < 4; ++nt) {
        v8f z = {};
        s[nt] = wmma(qa, kf[nt], z);
      }
      // ---- scale + relative-position bias + shift-window mask
      const int mb = mt * 16 + ((lane >> 4) << 3);
#pragma unroll
      for (int nt = 0; nt < 4; ++nt) {
        const int n_ = nt * 16 + n0;
#pragma unroll
        for (int r = 0; r < 8; ++r) {
          const int m_ = mb + r;
          float val;
          if (m_ < NT && n_ < NT) {
            const int pidx = m_ * NT + n_;
            const int tidx = (int)rpi[pidx];
            val = s[nt][r] * ATT_SCALE + bias_table[tidx * NHEADS + h] +
                  mask[(size_t)widx * NT * NT + pidx];
          } else {
            val = -1e30f;   // padded column -> prob 0
          }
          s[nt][r] = val;
        }
      }
      // ---- register-resident softmax (rows live in one 16-lane half)
      v8f mx = s[0];
#pragma unroll
      for (int nt = 1; nt < 4; ++nt)
#pragma unroll
        for (int r = 0; r < 8; ++r) mx[r] = fmaxf(mx[r], s[nt][r]);
#pragma unroll
      for (int off = 8; off >= 1; off >>= 1)
#pragma unroll
        for (int r = 0; r < 8; ++r)
          mx[r] = fmaxf(mx[r], __shfl_xor(mx[r], off, 16));
      v8f sum = {};
#pragma unroll
      for (int nt = 0; nt < 4; ++nt)
#pragma unroll
        for (int r = 0; r < 8; ++r) {
          float e = __expf(s[nt][r] - mx[r]);
          s[nt][r] = e;
          sum[r] += e;
        }
#pragma unroll
      for (int off = 8; off >= 1; off >>= 1)
#pragma unroll
        for (int r = 0; r < 8; ++r) sum[r] += __shfl_xor(sum[r], off, 16);
#pragma unroll
      for (int r = 0; r < 8; ++r) sum[r] = 1.0f / sum[r];
      // ---- probabilities to LDS (f16, row-major, stride NP)
#pragma unroll
      for (int nt = 0; nt < 4; ++nt) {
        const int n_ = nt * 16 + n0;
#pragma unroll
        for (int r = 0; r < 8; ++r)
          sAttnH[(mb + r) * NP + n_] = (_Float16)(s[nt][r] * sum[r]);
      }
    }

    asm volatile("s_wait_dscnt 0" ::: "memory");   // LDS RAW (same wave)

    // ---- context = attn @ V : K over 64 padded tokens = 2 WMMA steps
#pragma unroll
    for (int mt = 0; mt < 4; ++mt) {
#pragma unroll
      for (int dt = 0; dt < 2; ++dt) {
        v8f acc = {};
#pragma unroll
        for (int ks = 0; ks < 2; ++ks) {
          v16h af = a_frag_lds(sAttnH, mt * 16 + (lane & 15), NP,
                               ks * 32 + ((lane >> 4) << 4));
          const int dcol = dt * 16 + (lane & 15);
          v16h bf = b_frag_lds(sVtH + dcol * NP + ks * 32 + ((lane >> 4) << 4));
          acc = wmma(af, bf, acc);
        }
        const int col = h * HD + dt * 16 + (lane & 15);
        const int mb = mt * 16 + ((lane >> 4) << 3);
#pragma unroll
        for (int r = 0; r < 8; ++r)
          sX[(mb + r) * CDIM + col] = (_Float16)acc[r];
      }
    }
  }

  __syncthreads();

  // ---- fused output projection: out = x @ proj_w^T + proj_b
  // 4 M-tiles x 12 N-tiles, K=192 -> 6 WMMA steps; all 8 waves participate.
  for (int tile = wave; tile < 48; tile += 8) {
    const int mt = tile & 3;
    const int ot = tile >> 2;
    const int o = ot * 16 + (lane & 15);
    v8f acc = {};
#pragma unroll
    for (int ks = 0; ks < 6; ++ks) {
      v16h af = a_frag_lds(sX, mt * 16 + (lane & 15), CDIM,
                           ks * 32 + ((lane >> 4) << 4));
      // proj_w is (out,in) row-major: B[k=c][n=o] contiguous in c per row o
      v16h bf = b_frag_f32row(proj_w + (size_t)o * CDIM + ks * 32 +
                              ((lane >> 4) << 4));
      acc = wmma(af, bf, acc);
    }
    const float pb = proj_b[o];
    const int mb = mt * 16 + ((lane >> 4) << 3);
#pragma unroll
    for (int r = 0; r < 8; ++r) {
      const int m_ = mb + r;
      if (m_ < NT)
        out[((size_t)b * NT + m_) * CDIM + o] = acc[r] + pb;
    }
  }
}

extern "C" void kernel_launch(void* const* d_in, const int* in_sizes, int n_in,
                              void* d_out, int out_size, void* d_ws, size_t ws_size,
                              hipStream_t stream) {
  const float*     qkv = (const float*)d_in[0];
  const long long* rpi = (const long long*)d_in[1];   // int64 per reference
  const float*     msk = (const float*)d_in[2];
  const float*     bt  = (const float*)d_in[3];
  const float*     pw  = (const float*)d_in[4];
  const float*     pb  = (const float*)d_in[5];
  float*           out = (float*)d_out;
  (void)d_ws; (void)ws_size; (void)n_in; (void)out_size;

  const int B_ = in_sizes[0] / (NT * C3);          // 8192
  const int nw = in_sizes[2] / (NT * NT);          // 1024

  const size_t shmem = (size_t)(NHEADS * HD * NP +   // V^T   (f16)
                                NHEADS * NP * NP +   // probs (f16)
                                NP * CDIM) * 2;      // x     (f16)  = 96 KB

  swin_win_attn_fused<<<dim3(B_), dim3(256), shmem, stream>>>(
      qkv, rpi, msk, bt, pw, pb, out, nw);
}
